// VGG16_add_partial_558345748882
// MI455X (gfx1250) — 16-bit and 8-bit matrix operand order repaired and checked by reading, compiled, not run on hardware
//
#include <hip/hip_runtime.h>
#include <hip/hip_bf16.h>


typedef __attribute__((ext_vector_type(16))) _Float16 v16h;
typedef __attribute__((ext_vector_type(8)))  _Float16 v8h;
typedef __attribute__((ext_vector_type(8)))  float    v8f;
typedef __attribute__((ext_vector_type(8)))  int      v8i;
typedef __attribute__((ext_vector_type(4)))  int      v4i;
typedef __attribute__((ext_vector_type(2)))  int      v2i;
typedef __attribute__((ext_vector_type(8)))  unsigned char v8ub;

#define SC_B 256
#define SC_NPW 8

__device__ __forceinline__ float sc_quant(float v, float lo, float hi, float n, float rn) {
    float vc = fminf(fmaxf(v, lo), hi);
    return rintf(vc * n) * rn;
}

__global__ void sc_wquant(const float* __restrict__ w, _Float16* __restrict__ wq,
                          int CoTotal, int CoValid, int Ci, int Cp, int taps,
                          int Kpad, float n, float rn) {
    int tid = blockIdx.x * blockDim.x + threadIdx.x;
    int total = CoTotal * Kpad;
    if (tid >= total) return;
    int co = tid / Kpad;
    int k  = tid - co * Kpad;
    int dydx = k / Cp;
    int ci   = k - dydx * Cp;
    float v = 0.f;
    if (co < CoValid && dydx < taps && ci < Ci)
        v = sc_quant(w[((size_t)co * Ci + ci) * taps + dydx], -1.f, 1.f, n, rn);
    wq[tid] = (_Float16)v;
}

__global__ void sc_wquant_i8(const float* __restrict__ w, char* __restrict__ wq,
                             int Co, int Ci, int Cp, int Kpad, float n) {
    int tid = blockIdx.x * blockDim.x + threadIdx.x;
    int total = Co * Kpad;
    if (tid >= total) return;
    int co = tid / Kpad;
    int k  = tid - co * Kpad;
    int dydx = k / Cp;
    int ci   = k - dydx * Cp;
    int q = 0;
    if (dydx < 9 && ci < Ci) {
        float t = fminf(fmaxf(w[((size_t)co * Ci + ci) * 9 + dydx], -1.f), 1.f);
        q = (int)rintf(t * n);
    }
    wq[tid] = (char)q;
}

__global__ void sc_act0(const float* __restrict__ x, _Float16* __restrict__ a,
                        float n, float rn) {
    int tid = blockIdx.x * blockDim.x + threadIdx.x;
    const int Hp = 34, Wp = 34;
    int total = SC_B * Hp * Wp;
    if (tid >= total) return;
    int xx = tid % Wp;  int t1 = tid / Wp;
    int yy = t1 % Hp;   int b  = t1 / Hp;
    v8h lo = {}, hi = {};
    if (yy >= 1 && yy <= 32 && xx >= 1 && xx <= 32) {
        size_t src = ((size_t)b * 3) * 1024 + (size_t)(yy - 1) * 32 + (xx - 1);
#pragma unroll
        for (int ci = 0; ci < 3; ++ci)
            lo[ci] = (_Float16)sc_quant(x[src + (size_t)ci * 1024], 0.f, 1.f, n, rn);
    }
    _Float16* dst = a + (size_t)tid * 16;
    *(v8h*)(dst)     = lo;
    *(v8h*)(dst + 8) = hi;
}

__global__ void sc_actq(const float* __restrict__ in, void* __restrict__ outp,
                        int C, int H, int W, int pool, int pad, float n, float rn,
                        int u8out) {
    int Ho = pool ? (H >> 1) : H;
    int Wo = pool ? (W >> 1) : W;
    int Hp = Ho + 2 * pad, Wp = Wo + 2 * pad;
    int C8 = C >> 3;
    int tid = blockIdx.x * blockDim.x + threadIdx.x;
    int total = SC_B * Hp * Wp * C8;
    if (tid >= total) return;
    int c8 = tid % C8;  int t1 = tid / C8;
    int xx = t1 % Wp;   int t2 = t1 / Wp;
    int yy = t2 % Hp;   int b  = t2 / Hp;
    int y = yy - pad, x = xx - pad;
    bool inside = (y >= 0 && y < Ho && x >= 0 && x < Wo);
    v8f s = {};
    if (inside) {
        if (pool) {
            const float* p = in + (((size_t)b * H + 2 * y) * W + 2 * x) * C + c8 * 8;
            v8f a0 = *(const v8f*)(p);
            v8f a1 = *(const v8f*)(p + C);
            v8f a2 = *(const v8f*)(p + (size_t)W * C);
            v8f a3 = *(const v8f*)(p + (size_t)W * C + C);
            s = 0.25f * (a0 + a1 + a2 + a3);
        } else {
            s = *(const v8f*)(in + (((size_t)b * H + y) * W + x) * C + c8 * 8);
        }
    }
    if (u8out) {
        v8ub r = {};
        if (inside) {
#pragma unroll
            for (int t = 0; t < 8; ++t)
                r[t] = (unsigned char)(int)rintf(fminf(fmaxf(s[t], 0.f), 1.f) * n);
        }
        *(v8ub*)((unsigned char*)outp + (size_t)tid * 8) = r;
    } else {
        v8h r = {};
        if (inside) {
#pragma unroll
            for (int t = 0; t < 8; ++t)
                r[t] = (_Float16)sc_quant(s[t], 0.f, 1.f, n, rn);
        }
        *(v8h*)((_Float16*)outp + (size_t)tid * 8) = r;
    }
}

__global__ __launch_bounds__(128)
void sc_conv_wmma(const _Float16* __restrict__ act, const _Float16* __restrict__ wq,
                  const float* __restrict__ bn, float* __restrict__ out,
                  int Co, int H, int W, int lW, int lHW, int Cp, int lCp,
                  int Kpad, int Kg) {
    const int lane = threadIdx.x & 31;
    const int wave = threadIdx.x >> 5;
    const int half = lane >> 4;
    const int l16  = lane & 15;
    const int Hp = H + 2, Wp = W + 2;

    const int co_base = (blockIdx.x * 4 + wave) * 16;
    const _Float16* arow = wq + (size_t)(co_base + l16) * Kpad;

    size_t obase[SC_NPW];
    const _Float16* pbase[SC_NPW];
    const int pxbase = blockIdx.y * (16 * SC_NPW);
#pragma unroll
    for (int j = 0; j < SC_NPW; ++j) {
        int px  = pxbase + j * 16 + l16;
        int b   = px >> lHW;
        int rem = px & ((1 << lHW) - 1);
        int y   = rem >> lW;
        int x   = rem & ((1 << lW) - 1);
        pbase[j] = act + ((((size_t)b * Hp + y) * Wp + x) << lCp);
        obase[j] = (((size_t)b * H + y) * W + x) * Co;
    }

    v8f acc[SC_NPW] = {};

    for (int kk = 0; kk < Kpad; kk += 32) {
        __builtin_prefetch(arow + kk + 32, 0, 0);
        const v16h a = *(const v16h*)(arow + kk + (half << 4));

        const int kb0  = kk + (half << 4);
        const int dydx = kb0 >> lCp;
        const int cb   = kb0 & (Cp - 1);
        const int dy   = (dydx >= 6) ? 2 : ((dydx >= 3) ? 1 : 0);
        const int dx   = dydx - dy * 3;
        const int offs = ((dy * Wp + dx) << lCp) + cb;
        const bool live = (kb0 < Kg);
#pragma unroll
        for (int j = 0; j < SC_NPW; ++j) {
            v16h bf = {};
            if (live) bf = *(const v16h*)(pbase[j] + offs);
            acc[j] = __builtin_amdgcn_wmma_f32_16x16x32_f16(
                false, a, false, bf, (short)0, acc[j], false, false);
        }
    }

    const int comb = co_base + (half << 3);
    const v8f g8 = *(const v8f*)(bn + comb);
    const v8f b8 = *(const v8f*)(bn + Co + comb);
    const v8f m8 = *(const v8f*)(bn + 2 * Co + comb);
    const v8f v8_ = *(const v8f*)(bn + 3 * Co + comb);
    v8f s8;
#pragma unroll
    for (int v = 0; v < 8; ++v) s8[v] = g8[v] * __frsqrt_rn(v8_[v] + 1e-5f);
#pragma unroll
    for (int j = 0; j < SC_NPW; ++j) {
        v8f r;
#pragma unroll
        for (int v = 0; v < 8; ++v) r[v] = (acc[j][v] - m8[v]) * s8[v] + b8[v];
        *(v8f*)(out + obase[j] + comb) = r;
    }
}

__global__ __launch_bounds__(128)
void sc_conv_wmma_i8(const unsigned char* __restrict__ act, const char* __restrict__ wq,
                     const float* __restrict__ bn, float* __restrict__ out,
                     int Co, int H, int W, int lW, int lHW, int Cp, int lCp,
                     int Kpad, float invn2) {
    const int lane = threadIdx.x & 31;
    const int wave = threadIdx.x >> 5;
    const int half = lane >> 4;
    const int l16  = lane & 15;
    const int Hp = H + 2, Wp = W + 2;

    const int co_base = (blockIdx.x * 4 + wave) * 16;
    const char* arow = wq + (size_t)(co_base + l16) * Kpad;

    size_t obase[SC_NPW];
    const unsigned char* pbase[SC_NPW];
    const int pxbase = blockIdx.y * (16 * SC_NPW);
#pragma unroll
    for (int j = 0; j < SC_NPW; ++j) {
        int px  = pxbase + j * 16 + l16;
        int b   = px >> lHW;
        int rem = px & ((1 << lHW) - 1);
        int y   = rem >> lW;
        int x   = rem & ((1 << lW) - 1);
        pbase[j] = act + ((((size_t)b * Hp + y) * Wp + x) << lCp);
        obase[j] = (((size_t)b * H + y) * W + x) * Co;
    }

    v8i acc[SC_NPW] = {};

    for (int kk = 0; kk < Kpad; kk += 64) {
        const int ka0 = kk + (half << 4);
        __builtin_prefetch(arow + kk + 64, 0, 0);
        const v4i a03 = *(const v4i*)(arow + ka0);
        const v4i a47 = *(const v4i*)(arow + ka0 + 32);
        const v8i a = {a03[0], a03[1], a03[2], a03[3], a47[0], a47[1], a47[2], a47[3]};

        const int kb0a = kk + (half << 4);
        const int kb0b = kb0a + 32;
        const int dydx0 = kb0a >> lCp, cb0 = kb0a & (Cp - 1);
        const int dydx1 = kb0b >> lCp, cb1 = kb0b & (Cp - 1);
        const int dy0 = (dydx0 >= 6) ? 2 : ((dydx0 >= 3) ? 1 : 0);
        const int dy1 = (dydx1 >= 6) ? 2 : ((dydx1 >= 3) ? 1 : 0);
        const int dx0 = dydx0 - dy0 * 3;
        const int dx1 = dydx1 - dy1 * 3;
        const int offs0 = ((dy0 * Wp + dx0) << lCp) + cb0;
        const int offs1 = ((dy1 * Wp + dx1) << lCp) + cb1;
#pragma unroll
        for (int j = 0; j < SC_NPW; ++j) {
            const v4i b0 = *(const v4i*)(pbase[j] + offs0);
            const v4i b1 = *(const v4i*)(pbase[j] + offs1);
            const v8i bf = {b0[0], b0[1], b0[2], b0[3], b1[0], b1[1], b1[2], b1[3]};
            acc[j] = __builtin_amdgcn_wmma_i32_16x16x64_iu8(
                true  , a, false  , bf, acc[j], false, false);
        }
    }

    const int comb = co_base + (half << 3);
    const v8f g8 = *(const v8f*)(bn + comb);
    const v8f b8 = *(const v8f*)(bn + Co + comb);
    const v8f m8 = *(const v8f*)(bn + 2 * Co + comb);
    const v8f v8_ = *(const v8f*)(bn + 3 * Co + comb);
    v8f s8;
#pragma unroll
    for (int v = 0; v < 8; ++v) s8[v] = g8[v] * __frsqrt_rn(v8_[v] + 1e-5f);
#pragma unroll
    for (int j = 0; j < SC_NPW; ++j) {
        v8f r;
#pragma unroll
        for (int v = 0; v < 8; ++v)
            r[v] = ((float)acc[j][v] * invn2 - m8[v]) * s8[v] + b8[v];
        *(v8f*)(out + obase[j] + comb) = r;
    }
}

__global__ __launch_bounds__(32)
void sc_fc_wmma(const _Float16* __restrict__ fcin, const _Float16* __restrict__ wq,
                const float* __restrict__ bn, float* __restrict__ out) {
    const int lane = threadIdx.x;
    const int half = lane >> 4;
    const int l16  = lane & 15;
    const int bcol = blockIdx.x * 16 + l16;
    const _Float16* arow = wq + (size_t)l16 * 512;
    const _Float16* brow = fcin + (size_t)bcol * 512;

    v8f acc = {};
    for (int kk = 0; kk < 512; kk += 32) {
        const v16h a = *(const v16h*)(arow + kk + (half << 4));
        const v16h bf = *(const v16h*)(brow + kk + (half << 4));
        acc = __builtin_amdgcn_wmma_f32_16x16x32_f16(
            false, a, false, bf, (short)0, acc, false, false);
    }

#pragma unroll
    for (int v = 0; v < 8; ++v) {
        int com = (half << 3) + v;
        if (com < 10) {
            float g  = bn[com];
            float be = bn[10 + com];
            float mm = bn[20 + com];
            float vv = bn[30 + com];
            float s  = g * __frsqrt_rn(vv + 1e-5f);
            out[(size_t)bcol * 10 + com] = (acc[v] - mm) * s + be;
        }
    }
}

extern "C" void kernel_launch(void* const* d_in, const int* in_sizes, int n_in,
                              void* d_out, int out_size, void* d_ws, size_t ws_size,
                              hipStream_t stream) {
    (void)in_sizes; (void)n_in; (void)out_size; (void)ws_size;

    static const int CI[13]   = {3, 64, 64, 128, 128, 256, 256, 256, 512, 512, 512, 512, 512};
    static const int CO[13]   = {64, 64, 128, 128, 256, 256, 256, 512, 512, 512, 512, 512, 512};
    static const int ERRS[13] = {7, 6, 7, 6, 7, 7, 6, 7, 7, 6, 7, 7, 6};
    static const int POOL[13] = {0, 1, 0, 1, 0, 0, 1, 0, 0, 1, 0, 0, 1};

    char* ws = (char*)d_ws;
    void*  wqbuf = (void*)(ws);
    float* convs = (float*)(ws + ((size_t)8  << 20));
    void*  actA  = (void*)(ws + ((size_t)72 << 20));
    void*  actB  = (void*)(ws + ((size_t)112 << 20));

    const float* x = (const float*)d_in[0];

    {
        int total = SC_B * 34 * 34;
        float n = (float)(1 << ERRS[0]);
        sc_act0<<<(total + 255) / 256, 256, 0, stream>>>(x, (_Float16*)actA, n, 1.f / n);
    }

    void* cur = actA;
    void* nxt = actB;
    int H = 32, W = 32;

    for (int i = 0; i < 13; ++i) {
        const float* wi = (const float*)d_in[1 + i];
        const float* bn = (const float*)d_in[15 + i];
        int Ci = CI[i], Co = CO[i];
        int Cp = (Ci < 16) ? 16 : Ci;
        int lW  = __builtin_ctz(W);
        int lHW = __builtin_ctz(H * W);
        int lCp = __builtin_ctz(Cp);
        int Npx = SC_B * H * W;
        dim3 grid(Co / 64, Npx / (16 * SC_NPW));
        bool i8layer = (ERRS[i] == 6);

        if (i8layer) {
            int Kpad = 9 * Cp;
            sc_wquant_i8<<<(Co * Kpad + 255) / 256, 256, 0, stream>>>(
                wi, (char*)wqbuf, Co, Ci, Cp, Kpad, 64.f);
            sc_conv_wmma_i8<<<grid, 128, 0, stream>>>(
                (const unsigned char*)cur, (const char*)wqbuf, bn, convs,
                Co, H, W, lW, lHW, Cp, lCp, Kpad, 1.f / 4096.f);
        } else {
            int Kg = 9 * Cp;
            int Kpad = (Kg + 31) & ~31;
            float ni = (float)(1 << ERRS[i]);
            sc_wquant<<<(Co * Kpad + 255) / 256, 256, 0, stream>>>(
                wi, (_Float16*)wqbuf, Co, Co, Ci, Cp, 9, Kpad, ni, 1.f / ni);
            sc_conv_wmma<<<grid, 128, 0, stream>>>(
                (const _Float16*)cur, (const _Float16*)wqbuf, bn, convs,
                Co, H, W, lW, lHW, Cp, lCp, Kpad, Kg);
        }

        {
            int pool = POOL[i];
            int Ho = pool ? (H >> 1) : H;
            int Wo = pool ? (W >> 1) : W;
            int pad = (i == 12) ? 0 : 1;
            int nerr = (i < 12) ? ERRS[i + 1] : 7;
            int u8out = (i < 12 && ERRS[i + 1] == 6) ? 1 : 0;
            float nn = (float)(1 << nerr);
            int total = SC_B * (Ho + 2 * pad) * (Wo + 2 * pad) * (Co >> 3);
            sc_actq<<<(total + 255) / 256, 256, 0, stream>>>(
                convs, nxt, Co, H, W, pool, pad, nn, 1.f / nn, u8out);
            H = Ho; W = Wo;
            void* t = cur; cur = nxt; nxt = t;
        }
    }

    {
        const float* wfc  = (const float*)d_in[14];
        const float* bnfc = (const float*)d_in[28];
        float n7 = 128.f;
        sc_wquant<<<(16 * 512 + 255) / 256, 256, 0, stream>>>(
            wfc, (_Float16*)wqbuf, 16, 10, 512, 512, 1, 512, n7, 1.f / n7);
        sc_fc_wmma<<<dim3(SC_B / 16), 32, 0, stream>>>(
            (const _Float16*)cur, (const _Float16*)wqbuf, bnfc, (float*)d_out);
    }
}
